// ScaledDotProductAttention_11347303596482
// MI455X (gfx1250) — matrix operand order repaired and checked by reading, not compiled, not run on hardware
//
#include <hip/hip_runtime.h>

typedef __attribute__((ext_vector_type(16))) _Float16 v16h;
typedef __attribute__((ext_vector_type(8)))  _Float16 h8;
typedef __attribute__((ext_vector_type(4)))  _Float16 h4;
typedef __attribute__((ext_vector_type(8)))  float    v8f;
typedef __attribute__((ext_vector_type(4)))  float    f32x4;
typedef __attribute__((ext_vector_type(4)))  unsigned int u32x4;

#define SEQ    2048
#define DHEAD  64
#define BHE    32           // B*H
#define WAVES  8
#define WSZ    32
#define KCHUNK 64

#define SHUF16(a, b) __builtin_shufflevector(a, b, 0,1,2,3,4,5,6,7,8,9,10,11,12,13,14,15)
#define WMMA_F16(A, B, C) \
    __builtin_amdgcn_wmma_f32_16x16x32_f16(false, (A), false, (B), (short)0, (C), false, false)
// raw v_exp_f32 (2^x), no libm denormal fixup -- softmax never needs results < 2^-126
#define EXP2(x) __builtin_amdgcn_exp2f(x)

// ------------------------------------------------------------------
// One-shot pre-pass #1: K f32 -> f16, row major [bh][key][d]
// ------------------------------------------------------------------
__global__ __launch_bounds__(256)
void cvt_f32_to_f16(const float* __restrict__ src, _Float16* __restrict__ dst)
{
    const size_t i = (size_t)blockIdx.x * 256 + threadIdx.x;   // one f32x4 per thread
    const f32x4 w = ((const f32x4*)src)[i];
    h4 h;
    h[0] = (_Float16)w[0]; h[1] = (_Float16)w[1];
    h[2] = (_Float16)w[2]; h[3] = (_Float16)w[3];
    ((h4*)dst)[i] = h;
}

// ------------------------------------------------------------------
// One-shot pre-pass #2: V f32 [bh][key][d] -> f16 transposed [bh][d][key]
// ------------------------------------------------------------------
__global__ __launch_bounds__(256)
void transpose_v_f16(const float* __restrict__ V, _Float16* __restrict__ VT)
{
    __shared__ float t[32][33];
    const int tx = threadIdx.x, ty = threadIdx.y;
    const int key0 = blockIdx.x * 32;
    const int d0   = blockIdx.y * 32;
    const int bh   = blockIdx.z;
    const float*    Vh  = V  + (size_t)bh * SEQ * DHEAD;
    _Float16*       VTh = VT + (size_t)bh * DHEAD * SEQ;
    #pragma unroll
    for (int j = 0; j < 4; ++j)
        t[ty + j * 8][tx] = Vh[(size_t)(key0 + ty + j * 8) * DHEAD + d0 + tx];
    __syncthreads();
    #pragma unroll
    for (int j = 0; j < 4; ++j)
        VTh[(size_t)(d0 + ty + j * 8) * SEQ + key0 + tx] = (_Float16)t[tx][ty + j * 8];
}

// ------------------------------------------------------------------
// Main flash-attention kernel (fast path).
// Transposed dataflow:  S^T = K * Q^T   and   O^T = V^T * P^T,
// so softmax row stats are per-lane scalars (q lives in the lane dim).
// Q is pre-scaled by (1/sqrt(d)) * log2(e): scores are base-2 logits and
// softmax uses raw v_exp_f32 (exp2).  Masked scores are allowed into the
// running max (harmless); their weights are zeroed after exp with
// p = fma(-maskf, e, e), maskf via v_cvt_f32_ubyteN.
// 8 waves / block share double-buffered 64-key K / V^T tiles in LDS.
// ------------------------------------------------------------------
__global__ __launch_bounds__(WAVES * WSZ)
void sdpa_fa_wmma_lds(const float* __restrict__ Qg, const _Float16* __restrict__ K16,
                      const _Float16* __restrict__ VT16, const unsigned char* __restrict__ Mg,
                      float* __restrict__ Og)
{
    __shared__ _Float16 ldsK [2][KCHUNK * DHEAD];   // [key_local][d]
    __shared__ _Float16 ldsVT[2][DHEAD * KCHUNK];   // [d][key_local]
    __shared__ _Float16 ldsP [WAVES][32 * 16];      // P^T repack: [q][key_local]

    const int tid   = threadIdx.x;
    const int lane  = tid & (WSZ - 1);
    const int wid   = tid >> 5;
    const int row16 = lane & 15;
    const int half  = lane >> 4;

    const int qblks = SEQ / 16 / WAVES;             // 16
    const int bh    = blockIdx.x / qblks;
    const int qblk  = blockIdx.x % qblks;
    const int q0w   = (qblk * WAVES + wid) * 16;

    const float*    Qh  = Qg   + (size_t)bh * SEQ * DHEAD;
    const _Float16* Kh  = K16  + (size_t)bh * SEQ * DHEAD;
    const _Float16* VTh = VT16 + (size_t)bh * DHEAD * SEQ;
    float*          Oh  = Og   + (size_t)bh * SEQ * DHEAD;

    // (1/sqrt(64)) * log2(e), folded into Q at f16-conversion time.
    const float qscale = 0.18033688011112042f;

    // Q^T B-fragments (32 d x 16 q per d-chunk): lane = q column, element = d.
    v16h qt[2];
    #pragma unroll
    for (int dc = 0; dc < 2; ++dc)
        #pragma unroll
        for (int i = 0; i < 16; ++i)
            qt[dc][i] = (_Float16)(Qh[(size_t)(q0w + row16) * DHEAD + dc * 32 + (i & 7) + 16 * (i >> 3) + 8 * half] * qscale);

    v8f acc[4] = {};              // O^T C-tiles: tile t -> d = 16t + r + 8*half, q = lane&15
    float m_run = -1e30f, l_run = 0.0f;

    // cooperative stage of one 64-key chunk into LDS buffer `buf`
    auto stage = [&](int key0, int buf) {
        const u32x4* sk = (const u32x4*)(Kh + (size_t)key0 * DHEAD);   // 8 KB contiguous
        u32x4* dk = (u32x4*)ldsK[buf];
        dk[tid]       = sk[tid];
        dk[tid + 256] = sk[tid + 256];
        u32x4* dv = (u32x4*)ldsVT[buf];
        int i = tid;              // 512 x 16B: row = i>>3 (d), seg = i&7
        dv[i] = *(const u32x4*)(VTh + (size_t)(i >> 3) * SEQ + key0 + (i & 7) * 8);
        i = tid + 256;
        dv[i] = *(const u32x4*)(VTh + (size_t)(i >> 3) * SEQ + key0 + (i & 7) * 8);
    };

    stage(0, 0);
    __syncthreads();

    for (int c = 0; c < SEQ / KCHUNK; ++c) {
        const int kbase = c * KCHUNK;
        const int buf   = c & 1;
        if (c + 1 < SEQ / KCHUNK) {
            stage(kbase + KCHUNK, buf ^ 1);
            if (c + 2 < SEQ / KCHUNK)   // prime L2 one more chunk ahead
                __builtin_prefetch(Kh + (size_t)(kbase + 2 * KCHUNK) * DHEAD + tid * 16, 0, 1);
        }

        #pragma unroll
        for (int sc = 0; sc < 2; ++sc) {
            const int c0 = sc * 32;

            // ---- S^T = K * Q^T : A = K fragment (lane = key row, elem = d) ----
            const _Float16* lk = ldsK[buf];
            v8f s[2];
            #pragma unroll
            for (int kt = 0; kt < 2; ++kt) {
                const _Float16* krow = lk + (size_t)(c0 + kt * 16 + row16) * DHEAD;
                h8 a00 = *(const h8*)(krow +      8 * half);
                h8 a01 = *(const h8*)(krow + 16 + 8 * half);
                h8 a10 = *(const h8*)(krow + 32 + 8 * half);
                h8 a11 = *(const h8*)(krow + 48 + 8 * half);
                v16h ka0 = SHUF16(a00, a01);     // d-chunk 0
                v16h ka1 = SHUF16(a10, a11);     // d-chunk 1
                v8f z = {};
                z = WMMA_F16(ka0, qt[0], z);
                z = WMMA_F16(ka1, qt[1], z);
                s[kt] = z;
            }

            // ---- mask words; key = kbase + c0 + 16*kt + r + 8*half ----
            const size_t mo = (size_t)(q0w + row16) * SEQ + kbase + c0 + 8 * half;
            const unsigned long long mb0 = *(const unsigned long long*)(Mg + mo);
            const unsigned long long mb1 = *(const unsigned long long*)(Mg + mo + 16);
            const unsigned int mw[2][2] = {
                { (unsigned int)mb0, (unsigned int)(mb0 >> 32) },
                { (unsigned int)mb1, (unsigned int)(mb1 >> 32) } };

            // ---- online softmax (base-2), per-lane stats, one shuffle each ----
            float cm = fmaxf(s[0][0], s[1][0]);
            #pragma unroll
            for (int r = 1; r < 8; ++r) cm = fmaxf(cm, fmaxf(s[0][r], s[1][r]));
            cm = fmaxf(cm, __shfl_xor(cm, 16, WSZ));
            const float mnew = fmaxf(m_run, cm);
            const float corr = EXP2(m_run - mnew);
            m_run = mnew;

            float p[2][8];
            float sl = 0.0f;
            #pragma unroll
            for (int kt = 0; kt < 2; ++kt)
                #pragma unroll
                for (int r = 0; r < 8; ++r) {
                    const float e  = EXP2(s[kt][r] - mnew);
                    const float mf = (float)((mw[kt][r >> 2] >> ((r & 3) * 8)) & 0xffu);
                    const float z  = __builtin_fmaf(-mf, e, e);   // e * (1 - mask)
                    p[kt][r] = z;
                    sl += z;
                }
            sl += __shfl_xor(sl, 16, WSZ);
            l_run = __builtin_fmaf(l_run, corr, sl);
            #pragma unroll
            for (int t = 0; t < 4; ++t)
                #pragma unroll
                for (int r = 0; r < 8; ++r) acc[t][r] *= corr;

            // ---- repack P^T (C layout) -> B fragment via per-wave LDS tile ----
            _Float16* pw = ldsP[wid];
            #pragma unroll
            for (int kt = 0; kt < 2; ++kt)
                #pragma unroll
                for (int r = 0; r < 8; ++r)
                    pw[(size_t)row16 * 32 + kt * 16 + r + 8 * half] = (_Float16)p[kt][r];
            __builtin_amdgcn_wave_barrier();     // DS ops from one wave are in-order
            h8 b0 = *(const h8*)(pw + row16 * 32 + 8 * half);
            h8 b1 = *(const h8*)(pw + row16 * 32 + 16 + 8 * half);
            const v16h pb = SHUF16(b0, b1);      // B: lane = q column, elem = key

            // ---- O^T += V^T * P^T : A = V^T fragment (lane = d row, elem = key) ----
            const _Float16* lv = ldsVT[buf];
            #pragma unroll
            for (int t = 0; t < 4; ++t) {
                const _Float16* vrow = lv + (size_t)(t * 16 + row16) * KCHUNK;
                h8 v0 = *(const h8*)(vrow + c0 +      8 * half);
                h8 v1 = *(const h8*)(vrow + c0 + 16 + 8 * half);
                const v16h va = SHUF16(v0, v1);
                acc[t] = WMMA_F16(va, pb, acc[t]);
            }
        }
        __syncthreads();
    }

    // ---- normalize and store: d = 16t + 8*half + r is contiguous -> b128 ----
    const float inv = 1.0f / l_run;
    float* orow = Oh + (size_t)(q0w + row16) * DHEAD + 8 * half;
    #pragma unroll
    for (int t = 0; t < 4; ++t) {
        f32x4 lo = { acc[t][0] * inv, acc[t][1] * inv, acc[t][2] * inv, acc[t][3] * inv };
        f32x4 hi = { acc[t][4] * inv, acc[t][5] * inv, acc[t][6] * inv, acc[t][7] * inv };
        *(f32x4*)(orow + 16 * t)     = lo;
        *(f32x4*)(orow + 16 * t + 4) = hi;
    }
}

// ------------------------------------------------------------------
// Fallback (workspace too small): direct-f32 kernel.
// ------------------------------------------------------------------
__global__ __launch_bounds__(WAVES * WSZ)
void sdpa_fa_direct(const float* __restrict__ Qg, const float* __restrict__ Kg,
                    const float* __restrict__ Vg, const unsigned char* __restrict__ Mg,
                    float* __restrict__ Og)
{
    __shared__ _Float16 p_lds[WAVES][16 * 32];

    const int lane  = threadIdx.x & (WSZ - 1);
    const int wid   = threadIdx.x >> 5;
    const int row16 = lane & 15;
    const int half  = lane >> 4;

    const int qblks = SEQ / 16 / WAVES;
    const int bh    = blockIdx.x / qblks;
    const int qblk  = blockIdx.x % qblks;
    const int q0    = (qblk * WAVES + wid) * 16;

    const size_t base = (size_t)bh * SEQ * DHEAD;
    const float* Qh = Qg + base;
    const float* Kh = Kg + base;
    const float* Vh = Vg + base;
    float*       Oh = Og + base;

    v16h qa0, qa1;
    {
        const float* qrow = Qh + (size_t)(q0 + row16) * DHEAD;
        #pragma unroll
        for (int j = 0; j < 16; ++j) {
            const int kk = (j & 7) + 16 * (j >> 3) + 8 * half;
            qa0[j] = (_Float16)qrow[kk];
            qa1[j] = (_Float16)qrow[32 + kk];
        }
    }

    v8f acc[4] = {};
    float mrow[8], lrow[8];
    #pragma unroll
    for (int r = 0; r < 8; ++r) { mrow[r] = -1e30f; lrow[r] = 0.0f; }
    const float scale = 0.125f;

    for (int key0 = 0; key0 < SEQ; key0 += 32) {
        v16h kb00, kb01, kb10, kb11;
        #pragma unroll
        for (int i = 0; i < 16; ++i) {
            const int kk = (i & 7) + 16 * (i >> 3) + 8 * half;
            const size_t r0 = (size_t)(key0 + row16)      * DHEAD;
            const size_t r1 = (size_t)(key0 + 16 + row16) * DHEAD;
            kb00[i] = (_Float16)Kh[r0 + kk];
            kb10[i] = (_Float16)Kh[r0 + 32 + kk];
            kb01[i] = (_Float16)Kh[r1 + kk];
            kb11[i] = (_Float16)Kh[r1 + 32 + kk];
        }
        v8f s0 = {}, s1 = {};
        s0 = WMMA_F16(qa0, kb00, s0);
        s0 = WMMA_F16(qa1, kb10, s0);
        s1 = WMMA_F16(qa0, kb01, s1);
        s1 = WMMA_F16(qa1, kb11, s1);

        float p0[8], p1[8];
        #pragma unroll
        for (int r = 0; r < 8; ++r) {
            const int    qr = q0 + r + 8 * half;
            const size_t mb = (size_t)qr * SEQ + key0 + (lane & 15);
            float x0 = s0[r] * scale, x1 = s1[r] * scale;
            if (Mg[mb])      x0 = -1e30f;
            if (Mg[mb + 16]) x1 = -1e30f;
            p0[r] = x0; p1[r] = x1;
        }
        #pragma unroll
        for (int r = 0; r < 8; ++r) {
            float cm = fmaxf(p0[r], p1[r]);
            #pragma unroll
            for (int off = 8; off >= 1; off >>= 1)
                cm = fmaxf(cm, __shfl_xor(cm, off, WSZ));
            const float mnew = fmaxf(mrow[r], cm);
            const float corr = __expf(mrow[r] - mnew);
            mrow[r] = mnew;
            const float e0 = __expf(p0[r] - mnew);
            const float e1 = __expf(p1[r] - mnew);
            p0[r] = e0; p1[r] = e1;
            float sl = e0 + e1;
            #pragma unroll
            for (int off = 8; off >= 1; off >>= 1)
                sl += __shfl_xor(sl, off, WSZ);
            lrow[r] = lrow[r] * corr + sl;
            #pragma unroll
            for (int t = 0; t < 4; ++t) acc[t][r] *= corr;
        }

        _Float16* pw = p_lds[wid];
        #pragma unroll
        for (int r = 0; r < 8; ++r) {
            const int m = r + 8 * half;
            pw[m * 32 + (lane & 15)]      = (_Float16)p0[r];
            pw[m * 32 + 16 + (lane & 15)] = (_Float16)p1[r];
        }
        __builtin_amdgcn_wave_barrier();
        v16h pa;
        #pragma unroll
        for (int j = 0; j < 16; ++j) {
            const int kk = (j & 7) + 16 * (j >> 3) + 8 * half;
            pa[j] = pw[row16 * 32 + kk];
        }

        #pragma unroll
        for (int t = 0; t < 4; ++t) {
            v16h vb;
            #pragma unroll
            for (int j = 0; j < 16; ++j) {
                const int kk = (j & 7) + 16 * (j >> 3) + 8 * half;
                vb[j] = (_Float16)Vh[(size_t)(key0 + kk) * DHEAD + t * 16 + row16];
            }
            acc[t] = WMMA_F16(pa, vb, acc[t]);
        }
    }

    #pragma unroll
    for (int r = 0; r < 8; ++r) {
        const float inv = 1.0f / lrow[r];
        const int   qr  = q0 + r + 8 * half;
        #pragma unroll
        for (int t = 0; t < 4; ++t)
            Oh[(size_t)qr * DHEAD + t * 16 + (lane & 15)] = acc[t][r] * inv;
    }
}

// ------------------------------------------------------------------
extern "C" void kernel_launch(void* const* d_in, const int* in_sizes, int n_in,
                              void* d_out, int out_size, void* d_ws, size_t ws_size,
                              hipStream_t stream) {
    (void)in_sizes; (void)n_in; (void)out_size;
    const float*         Q = (const float*)d_in[0];
    const float*         K = (const float*)d_in[1];
    const float*         V = (const float*)d_in[2];
    const unsigned char* M = (const unsigned char*)d_in[3];
    float*               O = (float*)d_out;

    const size_t elems = (size_t)BHE * SEQ * DHEAD;          // 4,194,304
    const size_t need  = 2 * elems * sizeof(unsigned short); // K16 + VT16 = 16 MiB
    const int    blocks = BHE * (SEQ / 16 / WAVES);          // 512

    if (ws_size >= need) {
        _Float16* K16  = (_Float16*)d_ws;
        _Float16* VT16 = (_Float16*)((char*)d_ws + elems * sizeof(unsigned short));
        cvt_f32_to_f16<<<(int)(elems / 4 / 256), 256, 0, stream>>>(K, K16);
        transpose_v_f16<<<dim3(SEQ / 32, DHEAD / 32, BHE), dim3(32, 8), 0, stream>>>(V, VT16);
        sdpa_fa_wmma_lds<<<blocks, WAVES * WSZ, 0, stream>>>(Q, K16, VT16, M, O);
    } else {
        sdpa_fa_direct<<<blocks, WAVES * WSZ, 0, stream>>>(Q, K, V, M, O);
    }
}
